// DecisionTransformer_2491081031767
// MI455X (gfx1250) — matrix operand order repaired and checked by reading, not compiled, not run on hardware
//
#include <hip/hip_runtime.h>

// ---------------- problem constants ----------------
#define B_   8
#define T_   512
#define OBS_ 128
#define ACT_ 32
#define H_   768
#define NH_  12
#define L_   6
#define D_   64
#define S_   1536          // 3*T
#define H4_  3072
#define M_MAIN (B_*S_)     // 12288
#define M_HEAD (B_*T_)     // 4096

typedef __bf16 bf16;
typedef __attribute__((ext_vector_type(8)))  __bf16 bf16x8;
typedef __attribute__((ext_vector_type(16))) __bf16 v16bf;
typedef __attribute__((ext_vector_type(8)))  float  v8f;
typedef unsigned int u32x4 __attribute__((ext_vector_type(4)));
typedef int          i32x4 __attribute__((ext_vector_type(4)));
typedef int          i32x8 __attribute__((ext_vector_type(8)));

// ---------------- WMMA helpers (CDNA5 16x16x32 bf16, f32 accum) ----------------
__device__ __forceinline__ v8f wmma_bf16(v16bf a, v16bf b, v8f c) {
  return __builtin_amdgcn_wmma_f32_16x16x32_bf16(false, a, false, b, (short)0, c,
                                                 false, false);
}

// A fragment: 16x32 tile, row-major in LDS (ld = row stride, elements).
// Lanes 0-15 hold rows m with K {kb..kb+7, 16+kb..16+kb+7}, kb = 8*(lane>=16).
__device__ __forceinline__ v16bf frag_a16x32(const bf16* base, int ld) {
  int lane = threadIdx.x & 31;
  int m  = lane & 15;
  int kb = (lane >> 4) << 3;
  const bf16* p = base + m * ld + kb;
  bf16x8 lo = *(const bf16x8*)(p);
  bf16x8 hi = *(const bf16x8*)(p + 16);
  return __builtin_shufflevector(lo, hi, 0,1,2,3,4,5,6,7,8,9,10,11,12,13,14,15);
}

// B fragment: 32x16 (KxN) tile stored TRANSPOSED in LDS as [n][k].
// Lane n = lane&15, K {kb..kb+7, 16+kb..16+kb+7}, kb = 8*(lane>=16).
__device__ __forceinline__ v16bf frag_b32x16T(const bf16* baseT, int ld) {
  int lane = threadIdx.x & 31;
  int n  = lane & 15;
  int kb = (lane >> 4) << 3;
  const bf16* p = baseT + n * ld + kb;
  bf16x8 lo = *(const bf16x8*)(p);
  bf16x8 hi = *(const bf16x8*)(p + 16);
  return __builtin_shufflevector(lo, hi, 0,1,2,3,4,5,6,7,8,9,10,11,12,13,14,15);
}

__device__ __forceinline__ float gelu_f(float x) {
  float x3 = x * x * x;
  return 0.5f * x * (1.0f + tanhf(0.7978845608028654f * (x + 0.044715f * x3)));
}

// ---------------- Tensor Data Mover: 2-D tile -> LDS ----------------
// Loads a tile of tile_dim0=32 bf16 x tile_dim1=128 rows starting at `gtile`
// (row stride = kstride elements). Rows >= rows_rem are zero-filled (OOB).
// LDS padding: 4 DWORDs appended per 16-DWORD (64B) row -> LDS row stride 40 bf16.
// Descriptor fields per CDNA5 ISA 8.3-8.6.
__device__ __forceinline__ void tdm_load_tile(unsigned lds_addr, const bf16* gtile,
                                              int rows_rem, int kstride) {
  unsigned long long ga = (unsigned long long)(const void*)gtile;
  unsigned td0 = (unsigned)kstride;                    // tensor_dim0 (elements)
  unsigned td1 = (unsigned)(rows_rem > 0 ? rows_rem : 0); // tensor_dim1 (rows)
  u32x4 g0;
  g0[0] = 1u;                                          // count=1, user desc
  g0[1] = lds_addr;                                    // lds_addr
  g0[2] = (unsigned)ga;                                // global_addr[31:0]
  g0[3] = ((unsigned)(ga >> 32) & 0x01FFFFFFu) | (2u << 30); // addr[56:32]|type=2
  i32x8 g1;
  g1[0] = (int)((1u << 16)      // data_size = 1 (2 bytes)
              | (1u << 20)      // pad_enable
              | (3u << 22)      // pad_interval: 16 DWORDs (one 64B row)
              | (3u << 25));    // pad_amount: 4 DWORDs (16B)
  g1[1] = (int)((td0 & 0xFFFFu) << 16);                           // tensor_dim0 lo16
  g1[2] = (int)(((td0 >> 16) & 0xFFFFu) | ((td1 & 0xFFFFu) << 16)); // dim0 hi | dim1 lo
  g1[3] = (int)(((td1 >> 16) & 0xFFFFu) | (32u << 16));           // dim1 hi | tile_dim0=32
  g1[4] = (int)128u;                                              // tile_dim1=128, tile_dim2=0
  g1[5] = (int)td0;                                               // dim0_stride lo32
  g1[6] = 0;                                                      // dim0_stride hi16, dim1_stride lo
  g1[7] = 0;
  i32x4 z4; z4[0] = 0; z4[1] = 0; z4[2] = 0; z4[3] = 0;           // 2-D: groups 2/3 unused
#if defined(__clang_major__) && (__clang_major__ >= 23)
  i32x8 z8; z8[0]=0; z8[1]=0; z8[2]=0; z8[3]=0; z8[4]=0; z8[5]=0; z8[6]=0; z8[7]=0;
  __builtin_amdgcn_tensor_load_to_lds(g0, g1, z4, z4, z8, 0);
#else
  __builtin_amdgcn_tensor_load_to_lds(g0, g1, z4, z4, 0);
#endif
}

#define LDS_OFF(p) ((unsigned)(unsigned long long)(const void*)(p))

// ---------------- generic WMMA GEMM (TDM double-buffered) ----------------
// out[M,N] = act( A[M,K](bf16) @ WT[N,K](bf16, pre-transposed) + bias[N] (+ res) )
// 256 threads = 8 waves. Block tile 128x128, k-step 32. Wave tile 32x64 (2x4 wmma).
template<bool GELU, bool RES, bool OUTBF16>
__global__ __launch_bounds__(256)
void dt_gemm_wmma(const bf16* __restrict__ A, const bf16* __restrict__ WT,
                  const float* __restrict__ bias, const float* __restrict__ res,
                  float* __restrict__ outf, bf16* __restrict__ outb,
                  int M, int N, int K)
{
  __shared__ alignas(16) bf16 As[2][128 * 40];   // [m][k], stride 40 (TDM pad)
  __shared__ alignas(16) bf16 Bs[2][128 * 40];   // [n][k], stride 40

  const int tid  = threadIdx.x;
  const int lane = tid & 31;
  const int wid  = tid >> 5;
  const int m0 = blockIdx.y * 128;
  const int n0 = blockIdx.x * 128;
  const int wm = (wid & 3) * 32;
  const int wn = (wid >> 2) * 64;

  v8f acc[2][4];
#pragma unroll
  for (int mt = 0; mt < 2; ++mt)
#pragma unroll
    for (int nt = 0; nt < 4; ++nt)
#pragma unroll
      for (int i = 0; i < 8; ++i) acc[mt][nt][i] = 0.0f;

  const int nk = K >> 5;
  if (wid == 0) {  // wave 0 drives the Tensor Data Mover
    tdm_load_tile(LDS_OFF(&As[0][0]), A  + (size_t)m0 * K, M - m0, K);
    tdm_load_tile(LDS_OFF(&Bs[0][0]), WT + (size_t)n0 * K, N - n0, K);
  }

  for (int i = 0; i < nk; ++i) {
    const int buf = i & 1;
    if (wid == 0) {
      if (i + 1 < nk) {   // prefetch next k-tile into the other buffer
        tdm_load_tile(LDS_OFF(&As[buf ^ 1][0]),
                      A  + (size_t)m0 * K + (i + 1) * 32, M - m0, K);
        tdm_load_tile(LDS_OFF(&Bs[buf ^ 1][0]),
                      WT + (size_t)n0 * K + (i + 1) * 32, N - n0, K);
        __builtin_amdgcn_s_wait_tensorcnt(2);   // current buffer's 2 ops retired
      } else {
        __builtin_amdgcn_s_wait_tensorcnt(0);
      }
    }
    __syncthreads();

    v16bf a0 = frag_a16x32(&As[buf][(wm +  0) * 40], 40);
    v16bf a1 = frag_a16x32(&As[buf][(wm + 16) * 40], 40);
#pragma unroll
    for (int nt = 0; nt < 4; ++nt) {
      v16bf bfr = frag_b32x16T(&Bs[buf][(wn + nt * 16) * 40], 40);
      acc[0][nt] = wmma_bf16(a0, bfr, acc[0][nt]);
      acc[1][nt] = wmma_bf16(a1, bfr, acc[1][nt]);
    }
    __syncthreads();
  }

  // epilogue: C layout -> (m = i + 8*(lane>=16), n = lane&15)
  const int mrow = (lane >> 4) << 3;
  const int ncol = lane & 15;
#pragma unroll
  for (int mt = 0; mt < 2; ++mt)
#pragma unroll
    for (int nt = 0; nt < 4; ++nt)
#pragma unroll
      for (int i = 0; i < 8; ++i) {
        int row = m0 + wm + mt * 16 + mrow + i;
        int col = n0 + wn + nt * 16 + ncol;
        if (row < M && col < N) {
          float v = acc[mt][nt][i];
          if (bias) v += bias[col];
          if (RES)  v += res[(size_t)row * N + col];
          if (GELU) v = gelu_f(v);
          if (OUTBF16) outb[(size_t)row * N + col] = (bf16)v;
          else         outf[(size_t)row * N + col] = v;
        }
      }
}

// ---------------- weight convert + transpose: f32 [K][N] -> bf16 [N][K] ----------------
__global__ __launch_bounds__(256)
void dt_transpose_bf16(const float* __restrict__ w, bf16* __restrict__ wt,
                       int K, int N)
{
  __shared__ float tile[32][33];
  int k0 = blockIdx.y * 32, n0 = blockIdx.x * 32;
  int tx = threadIdx.x & 31, ty = threadIdx.x >> 5;   // 8 rows per pass
#pragma unroll
  for (int i = 0; i < 32; i += 8) {
    int k = k0 + ty + i, n = n0 + tx;
    tile[ty + i][tx] = (k < K && n < N) ? w[(size_t)k * N + n] : 0.0f;
  }
  __syncthreads();
#pragma unroll
  for (int i = 0; i < 32; i += 8) {
    int n = n0 + ty + i, k = k0 + tx;
    if (n < N && k < K) wt[(size_t)n * K + k] = (bf16)tile[tx][ty + i];
  }
}

// ---------------- causal flash attention ----------------
// grid (S/128, NH, B), 256 threads (8 waves). Each wave: 16 q-rows.
// Q,K,V,O: bf16 [B*S, H]; head h uses cols h*64..h*64+63.
__global__ __launch_bounds__(256)
void dt_attn(const bf16* __restrict__ Q, const bf16* __restrict__ Kg,
             const bf16* __restrict__ Vg, bf16* __restrict__ O)
{
  __shared__ alignas(16) bf16 Qs [128 * 64];     // [m][d]
  __shared__ alignas(16) bf16 Ks [128 * 64];     // [key][d]  (B layout [n=key][k=d])
  __shared__ alignas(16) bf16 VsT[ 64 * 128];    // [d][key]  (B layout [n=d][k=key])
  __shared__ alignas(16) bf16 Ps [8 * 16 * 128]; // per-wave [m][key]

  const int tid  = threadIdx.x;
  const int lane = tid & 31;
  const int wid  = tid >> 5;
  const int qt = blockIdx.x, h = blockIdx.y, b = blockIdx.z;
  const size_t rowbase = (size_t)b * S_;
  const int col0 = h * D_;

#pragma unroll
  for (int it = 0; it < 4; ++it) {
    int c  = it * 256 + tid;
    int r  = c >> 3;
    int cc = (c & 7) * 8;
    *(bf16x8*)(&Qs[r * 64 + cc]) =
        *(const bf16x8*)(Q + (rowbase + qt * 128 + r) * H_ + col0 + cc);
  }

  float m_i[8], l_i[8];
  v8f o_acc[4];
#pragma unroll
  for (int i = 0; i < 8; ++i) { m_i[i] = -__builtin_inff(); l_i[i] = 0.0f; }
#pragma unroll
  for (int nt = 0; nt < 4; ++nt)
#pragma unroll
    for (int i = 0; i < 8; ++i) o_acc[nt][i] = 0.0f;

  bf16* Pw = &Ps[wid * 16 * 128];
  const int mloc  = (lane >> 4) << 3;
  const int ncol  = lane & 15;
  const int qrow0 = qt * 128 + wid * 16;

  for (int kt = 0; kt <= qt; ++kt) {
    __syncthreads();
#pragma unroll
    for (int it = 0; it < 4; ++it) {
      int c  = it * 256 + tid;
      int r  = c >> 3;
      int cc = (c & 7) * 8;
      *(bf16x8*)(&Ks[r * 64 + cc]) =
          *(const bf16x8*)(Kg + (rowbase + kt * 128 + r) * H_ + col0 + cc);
      bf16x8 vv = *(const bf16x8*)(Vg + (rowbase + kt * 128 + r) * H_ + col0 + cc);
#pragma unroll
      for (int j = 0; j < 8; ++j) VsT[(cc + j) * 128 + r] = vv[j];
    }
    __syncthreads();

    v8f s_acc[8];
#pragma unroll
    for (int nt = 0; nt < 8; ++nt)
#pragma unroll
      for (int i = 0; i < 8; ++i) s_acc[nt][i] = 0.0f;
#pragma unroll
    for (int ks = 0; ks < 2; ++ks) {
      v16bf aq = frag_a16x32(&Qs[wid * 16 * 64 + ks * 32], 64);
#pragma unroll
      for (int nt = 0; nt < 8; ++nt) {
        v16bf bk = frag_b32x16T(&Ks[nt * 16 * 64 + ks * 32], 64);
        s_acc[nt] = wmma_bf16(aq, bk, s_acc[nt]);
      }
    }

    // online softmax per row (rows in vgpr index; n across 16-lane halves)
#pragma unroll
    for (int i = 0; i < 8; ++i) {
      int qr = qrow0 + mloc + i;
      float mx = -__builtin_inff();
#pragma unroll
      for (int nt = 0; nt < 8; ++nt) {
        int kc = kt * 128 + nt * 16 + ncol;
        float sv = s_acc[nt][i] * 0.125f;            // 1/sqrt(64)
        sv = (kc <= qr) ? sv : -__builtin_inff();    // causal mask
        s_acc[nt][i] = sv;
        mx = fmaxf(mx, sv);
      }
#pragma unroll
      for (int off = 1; off < 16; off <<= 1)
        mx = fmaxf(mx, __shfl_xor(mx, off, 32));
      float mnew = fmaxf(m_i[i], mx);
      float rs = 0.0f;
#pragma unroll
      for (int nt = 0; nt < 8; ++nt) {
        float pv = __expf(s_acc[nt][i] - mnew);
        s_acc[nt][i] = pv;
        rs += pv;
      }
#pragma unroll
      for (int off = 1; off < 16; off <<= 1)
        rs += __shfl_xor(rs, off, 32);
      float corr = __expf(m_i[i] - mnew);
      l_i[i] = l_i[i] * corr + rs;
      m_i[i] = mnew;
#pragma unroll
      for (int nt = 0; nt < 4; ++nt) o_acc[nt][i] *= corr;
#pragma unroll
      for (int nt = 0; nt < 8; ++nt)
        Pw[(mloc + i) * 128 + nt * 16 + ncol] = (bf16)s_acc[nt][i];
    }
    __syncthreads();

    // O += P V
#pragma unroll
    for (int ks = 0; ks < 4; ++ks) {
      v16bf ap = frag_a16x32(&Pw[ks * 32], 128);
#pragma unroll
      for (int ntd = 0; ntd < 4; ++ntd) {
        v16bf bv = frag_b32x16T(&VsT[ntd * 16 * 128 + ks * 32], 128);
        o_acc[ntd] = wmma_bf16(ap, bv, o_acc[ntd]);
      }
    }
  }

#pragma unroll
  for (int ntd = 0; ntd < 4; ++ntd)
#pragma unroll
    for (int i = 0; i < 8; ++i) {
      float v = o_acc[ntd][i] / l_i[i];
      int sg = qrow0 + mloc + i;
      O[(rowbase + sg) * H_ + col0 + ntd * 16 + ncol] = (bf16)v;
    }
}

// ---------------- embeddings ----------------
__global__ __launch_bounds__(256)
void dt_embed(const int* __restrict__ timesteps, const float* __restrict__ obs,
              const float* __restrict__ act, const float* __restrict__ rtg,
              const float* __restrict__ te_w,
              const float* __restrict__ rtg_w, const float* __restrict__ rtg_b,
              const float* __restrict__ obs_w, const float* __restrict__ obs_b,
              const float* __restrict__ act_w, const float* __restrict__ act_b,
              float* __restrict__ out)
{
  int bt = blockIdx.x;
  int bb = bt / T_, t = bt % T_;
  int ts = timesteps[bt];
  size_t base = ((size_t)bb * S_ + 3 * (size_t)t) * H_;
  float r0 = rtg[bt];
  for (int hh = threadIdx.x; hh < H_; hh += 256) {
    float te = te_w[(size_t)ts * H_ + hh];
    float rv = r0 * rtg_w[hh] + rtg_b[hh] + te;
    float ov = obs_b[hh] + te;
    for (int j = 0; j < OBS_; ++j)
      ov += obs[(size_t)bt * OBS_ + j] * obs_w[(size_t)j * H_ + hh];
    float av = act_b[hh] + te;
    for (int j = 0; j < ACT_; ++j)
      av += act[(size_t)bt * ACT_ + j] * act_w[(size_t)j * H_ + hh];
    out[base + hh]          = rv;
    out[base + H_ + hh]     = ov;
    out[base + 2 * H_ + hh] = av;
  }
}

// ---------------- layernorm (one block per row) ----------------
__global__ __launch_bounds__(256)
void dt_layernorm(const float* __restrict__ in, const float* __restrict__ g,
                  const float* __restrict__ bta, float* __restrict__ outf,
                  bf16* __restrict__ outb)
{
  __shared__ float red1[256];
  __shared__ float red2[256];
  int row = blockIdx.x;
  const float* x = in + (size_t)row * H_;
  float s = 0.0f, s2 = 0.0f;
  for (int i = threadIdx.x; i < H_; i += 256) {
    float v = x[i];
    s += v; s2 += v * v;
  }
  red1[threadIdx.x] = s; red2[threadIdx.x] = s2;
  __syncthreads();
  for (int o = 128; o > 0; o >>= 1) {
    if (threadIdx.x < o) {
      red1[threadIdx.x] += red1[threadIdx.x + o];
      red2[threadIdx.x] += red2[threadIdx.x + o];
    }
    __syncthreads();
  }
  float mean = red1[0] / H_;
  float var  = red2[0] / H_ - mean * mean;
  float rstd = rsqrtf(var + 1e-5f);
  for (int i = threadIdx.x; i < H_; i += 256) {
    float v = (x[i] - mean) * rstd * g[i] + bta[i];
    outf[(size_t)row * H_ + i] = v;
    outb[(size_t)row * H_ + i] = (bf16)v;
  }
}

// ---------------- gather obs/act token rows ----------------
__global__ void dt_gather(const float* __restrict__ x, bf16* __restrict__ hobs,
                          bf16* __restrict__ hact)
{
  int idx = blockIdx.x * 256 + threadIdx.x;
  if (idx >= M_HEAD * H_) return;
  int bt = idx / H_, hh = idx % H_;
  int bb = bt / T_, t = bt % T_;
  hobs[idx] = (bf16)x[((size_t)(bb * S_ + 3 * t + 1)) * H_ + hh];
  hact[idx] = (bf16)x[((size_t)(bb * S_ + 3 * t + 2)) * H_ + hh];
}

// ---------------- rtg head (H -> 1 dot) ----------------
__global__ __launch_bounds__(256)
void dt_rtg_head(const float* __restrict__ x, const float* __restrict__ w,
                 const float* __restrict__ bta, float* __restrict__ out)
{
  __shared__ float red[256];
  int bt = blockIdx.x;
  int bb = bt / T_, t = bt % T_;
  const float* row = x + ((size_t)(bb * S_ + 3 * t + 2)) * H_;
  float s = 0.0f;
  for (int i = threadIdx.x; i < H_; i += 256) s += row[i] * w[i];
  red[threadIdx.x] = s;
  __syncthreads();
  for (int o = 128; o > 0; o >>= 1) {
    if (threadIdx.x < o) red[threadIdx.x] += red[threadIdx.x + o];
    __syncthreads();
  }
  if (threadIdx.x == 0) out[bt] = red[0] + bta[0];
}

// ---------------- host orchestration ----------------
extern "C" void kernel_launch(void* const* d_in, const int* in_sizes, int n_in,
                              void* d_out, int out_size, void* d_ws, size_t ws_size,
                              hipStream_t stream)
{
  const int*   timesteps = (const int*)  d_in[0];
  const float* observ    = (const float*)d_in[1];
  const float* actions   = (const float*)d_in[2];
  const float* rtg       = (const float*)d_in[3];
  const float* te_w      = (const float*)d_in[4];
  const float* e_rtg_w   = (const float*)d_in[5];
  const float* e_rtg_b   = (const float*)d_in[6];
  const float* e_obs_w   = (const float*)d_in[7];
  const float* e_obs_b   = (const float*)d_in[8];
  const float* e_act_w   = (const float*)d_in[9];
  const float* e_act_b   = (const float*)d_in[10];
  const float* ln_g      = (const float*)d_in[11];
  const float* ln_b      = (const float*)d_in[12];
  const float* qw = (const float*)d_in[13]; const float* qb = (const float*)d_in[14];
  const float* kw = (const float*)d_in[15]; const float* kb = (const float*)d_in[16];
  const float* vw = (const float*)d_in[17]; const float* vb = (const float*)d_in[18];
  const float* pw = (const float*)d_in[19]; const float* pb = (const float*)d_in[20];
  const float* w1 = (const float*)d_in[21]; const float* b1 = (const float*)d_in[22];
  const float* w2 = (const float*)d_in[23]; const float* b2 = (const float*)d_in[24];
  const float* ln1_g = (const float*)d_in[25]; const float* ln1_b = (const float*)d_in[26];
  const float* ln2_g = (const float*)d_in[27]; const float* ln2_b = (const float*)d_in[28];
  const float* pr_rtg_w = (const float*)d_in[29]; const float* pr_rtg_b = (const float*)d_in[30];
  const float* pr_obs_w = (const float*)d_in[31]; const float* pr_obs_b = (const float*)d_in[32];
  const float* pr_act_w = (const float*)d_in[33]; const float* pr_act_b = (const float*)d_in[34];

  // workspace carve-up
  char* wsb = (char*)d_ws;
  size_t off = 0;
  auto carve = [&](size_t bytes) -> void* {
    void* p = wsb + off;
    off += (bytes + 255) & ~(size_t)255;
    return p;
  };
  float* tmpf  = (float*)carve((size_t)M_MAIN * H_  * 4);
  float* xf    = (float*)carve((size_t)M_MAIN * H_  * 4);
  bf16*  xb    = (bf16*) carve((size_t)M_MAIN * H_  * 2);
  bf16*  qbuf  = (bf16*) carve((size_t)M_MAIN * H_  * 2);
  bf16*  kbuf  = (bf16*) carve((size_t)M_MAIN * H_  * 2);
  bf16*  vbuf  = (bf16*) carve((size_t)M_MAIN * H_  * 2);
  bf16*  attnb = (bf16*) carve((size_t)M_MAIN * H_  * 2);
  bf16*  h1b   = (bf16*) carve((size_t)M_MAIN * H4_ * 2);
  bf16*  hobsb = (bf16*) carve((size_t)M_HEAD * H_  * 2);
  bf16*  hactb = (bf16*) carve((size_t)M_HEAD * H_  * 2);
  // per-layer transposed bf16 weights
  bf16*  wqt   = (bf16*) carve((size_t)H_  * H_  * 2);  // [768][768]
  bf16*  wkt   = (bf16*) carve((size_t)H_  * H_  * 2);
  bf16*  wvt   = (bf16*) carve((size_t)H_  * H_  * 2);
  bf16*  wpt   = (bf16*) carve((size_t)H_  * H_  * 2);
  bf16*  w1t   = (bf16*) carve((size_t)H4_ * H_  * 2);  // [3072][768]
  bf16*  w2t   = (bf16*) carve((size_t)H_  * H4_ * 2);  // [768][3072]
  bf16*  pot   = (bf16*) carve((size_t)OBS_* H_  * 2);  // [128][768]
  bf16*  pat   = (bf16*) carve((size_t)ACT_* H_  * 2);  // [32][768]
  (void)in_sizes; (void)n_in; (void)out_size; (void)ws_size;

  // embeddings + input LN
  dt_embed<<<M_HEAD, 256, 0, stream>>>(timesteps, observ, actions, rtg, te_w,
                                       e_rtg_w, e_rtg_b, e_obs_w, e_obs_b,
                                       e_act_w, e_act_b, tmpf);
  dt_layernorm<<<M_MAIN, 256, 0, stream>>>(tmpf, ln_g, ln_b, xf, xb);

  dim3 g768(H_ / 128, M_MAIN / 128);
  dim3 g3072(H4_ / 128, M_MAIN / 128);
  dim3 gattn(S_ / 128, NH_, B_);
  dim3 tHH(H_ / 32, H_ / 32);            // transpose grids
  dim3 tH_H4(H4_ / 32, H_ / 32);         // K=768,  N=3072
  dim3 tH4_H(H_ / 32, H4_ / 32);         // K=3072, N=768

  for (int l = 0; l < L_; ++l) {
    const float* qw_l = qw + (size_t)l * H_ * H_;  const float* qb_l = qb + (size_t)l * H_;
    const float* kw_l = kw + (size_t)l * H_ * H_;  const float* kb_l = kb + (size_t)l * H_;
    const float* vw_l = vw + (size_t)l * H_ * H_;  const float* vb_l = vb + (size_t)l * H_;
    const float* pw_l = pw + (size_t)l * H_ * H_;  const float* pb_l = pb + (size_t)l * H_;
    const float* w1_l = w1 + (size_t)l * H_ * H4_; const float* b1_l = b1 + (size_t)l * H4_;
    const float* w2_l = w2 + (size_t)l * H4_ * H_; const float* b2_l = b2 + (size_t)l * H_;
    const float* l1g = ln1_g + (size_t)l * H_;     const float* l1b = ln1_b + (size_t)l * H_;
    const float* l2g = ln2_g + (size_t)l * H_;     const float* l2b = ln2_b + (size_t)l * H_;

    // convert+transpose this layer's weights to bf16 [N][K]
    dt_transpose_bf16<<<tHH,   256, 0, stream>>>(qw_l, wqt, H_,  H_);
    dt_transpose_bf16<<<tHH,   256, 0, stream>>>(kw_l, wkt, H_,  H_);
    dt_transpose_bf16<<<tHH,   256, 0, stream>>>(vw_l, wvt, H_,  H_);
    dt_transpose_bf16<<<tHH,   256, 0, stream>>>(pw_l, wpt, H_,  H_);
    dt_transpose_bf16<<<tH_H4, 256, 0, stream>>>(w1_l, w1t, H_,  H4_);
    dt_transpose_bf16<<<tH4_H, 256, 0, stream>>>(w2_l, w2t, H4_, H_);

    // q, k, v projections -> bf16
    dt_gemm_wmma<false, false, true><<<g768, 256, 0, stream>>>(
        xb, wqt, qb_l, nullptr, nullptr, qbuf, M_MAIN, H_, H_);
    dt_gemm_wmma<false, false, true><<<g768, 256, 0, stream>>>(
        xb, wkt, kb_l, nullptr, nullptr, kbuf, M_MAIN, H_, H_);
    dt_gemm_wmma<false, false, true><<<g768, 256, 0, stream>>>(
        xb, wvt, vb_l, nullptr, nullptr, vbuf, M_MAIN, H_, H_);
    // attention
    dt_attn<<<gattn, 256, 0, stream>>>(qbuf, kbuf, vbuf, attnb);
    // output projection + residual -> f32, then LN1
    dt_gemm_wmma<false, true, false><<<g768, 256, 0, stream>>>(
        attnb, wpt, pb_l, xf, tmpf, nullptr, M_MAIN, H_, H_);
    dt_layernorm<<<M_MAIN, 256, 0, stream>>>(tmpf, l1g, l1b, xf, xb);
    // MLP: gelu(x @ w1 + b1) -> bf16 ; @ w2 + b2 + residual -> f32 ; LN2
    dt_gemm_wmma<true, false, true><<<g3072, 256, 0, stream>>>(
        xb, w1t, b1_l, nullptr, nullptr, h1b, M_MAIN, H4_, H_);
    dt_gemm_wmma<false, true, false><<<g768, 256, 0, stream>>>(
        h1b, w2t, b2_l, xf, tmpf, nullptr, M_MAIN, H_, H4_);
    dt_layernorm<<<M_MAIN, 256, 0, stream>>>(tmpf, l2g, l2b, xf, xb);
  }

  // heads: d_out = [rtg(B*T) | obs(B*T*128) | act(B*T*32)] f32
  dt_gather<<<(M_HEAD * H_ + 255) / 256, 256, 0, stream>>>(xf, hobsb, hactb);
  dt_transpose_bf16<<<dim3(OBS_ / 32, H_ / 32), 256, 0, stream>>>(pr_obs_w, pot, H_, OBS_);
  dt_transpose_bf16<<<dim3(1,        H_ / 32), 256, 0, stream>>>(pr_act_w, pat, H_, ACT_);
  float* out_rtg = (float*)d_out;
  float* out_obs = out_rtg + M_HEAD;
  float* out_act = out_obs + (size_t)M_HEAD * OBS_;
  dt_rtg_head<<<M_HEAD, 256, 0, stream>>>(xf, pr_rtg_w, pr_rtg_b, out_rtg);
  dim3 gobs(1, M_HEAD / 128);
  dt_gemm_wmma<false, false, false><<<gobs, 256, 0, stream>>>(
      hactb, pot, pr_obs_b, nullptr, out_obs, nullptr, M_HEAD, OBS_, H_);
  dim3 gact(1, M_HEAD / 128);
  dt_gemm_wmma<false, false, false><<<gact, 256, 0, stream>>>(
      hobsb, pat, pr_act_b, nullptr, out_act, nullptr, M_HEAD, ACT_, H_);
}
